// LanguageModel_2044404433406
// MI455X (gfx1250) — compile-verified, matrix operand order checked by reading, not run on hardware
//
#include <hip/hip_runtime.h>

// ---------------------------------------------------------------------------
// 2-layer LSTM-with-projection LM for MI455X (gfx1250, wave32, WMMA + TDM).
// bf16 weights pre-packed into WMMA-B fragment order; f32 accumulation via
// v_wmma_f32_16x16x32_bf16. Cell kernel: 1 workgroup per hidden tile, the
// Tensor Data Mover streams the block's 192KB weight slice into LDS (ih slab
// overlapped ahead of hh slab), 4 M-tile waves consume it via ds_load.
// ---------------------------------------------------------------------------

typedef __attribute__((ext_vector_type(16))) __bf16 v16bf;
typedef __attribute__((ext_vector_type(8)))  float  v8f;
typedef __attribute__((ext_vector_type(4)))  float  f32x4;
typedef __attribute__((ext_vector_type(4)))  unsigned int u32x4;
typedef __attribute__((ext_vector_type(8)))  int    i32x8;
typedef __attribute__((ext_vector_type(4)))  int    i32x4;

#define T_STEPS 256
#define BATCH   64
#define IN_DIM  512
#define H_DIM   1024

union FragU { v16bf v; u32x4 q[2]; };

// 16-bit A-matrix 16x32 layout (ISA 7.12.2):
//   lanes 0-15 : M=lane, K = {k0..k0+7} then {k0+16..k0+23}
//   lanes 16-31: M=lane-16, K = {k0+8..k0+15} then {k0+24..k0+31}
__device__ __forceinline__ v16bf load_a_frag(const __bf16* __restrict__ base,
                                             int row, int ldk, int k0, int lane) {
  FragU f;
  const __bf16* p = base + (size_t)row * ldk + k0 + ((lane >> 4) << 3);
  f.q[0] = *(const u32x4*)(p);
  f.q[1] = *(const u32x4*)(p + 16);
  return f.v;
}

__device__ __forceinline__ float sigmoid_f(float x) {
  return 1.0f / (1.0f + __expf(-x));
}
__device__ __forceinline__ float tanh_f(float x) {
  float t = __expf(-2.0f * fabsf(x));
  float r = (1.0f - t) / (1.0f + t);
  return (x < 0.0f) ? -r : r;
}

// ---------------------------------------------------------------------------
// TDM 2-D tile load: 4 rows (gates) x dim0 8-byte elements, row stride
// stride0 (8-byte units), into LDS at byte offset lds_off.
// D# layout per CDNA5 ISA.
// Toolchain uses the 6-arg builtin: (g0, g1, g2, g3, g_extra, cpol).
// ---------------------------------------------------------------------------
__device__ __forceinline__ void tdm_load_2d(unsigned lds_off, const void* gaddr,
                                            unsigned dim0, unsigned stride0) {
  unsigned long long ga = (unsigned long long)(uintptr_t)gaddr;
  u32x4 g0;
  g0.x = 1u;                                     // count=1 (valid user D#)
  g0.y = lds_off;                                // lds_addr [63:32]
  g0.z = (unsigned)(ga & 0xffffffffu);           // global_addr [95:64]
  g0.w = (unsigned)((ga >> 32) & 0x01ffffffu)    // global_addr [120:96]
       | (2u << 30);                             // type=2 ("image")
  i32x8 g1;
  g1[0] = (3 << 16);                             // data_size=3 (8B); wg_mask=0
  g1[1] = (int)((dim0 & 0xffffu) << 16);         // tensor_dim0 [15:0]
  g1[2] = (int)((dim0 >> 16) & 0xffffu)          // tensor_dim0 [31:16]
        | (4 << 16);                             // tensor_dim1 = 4 (low 16)
  g1[3] = (int)((dim0 & 0xffffu) << 16);         // tile_dim0
  g1[4] = 4;                                     // tile_dim1 = 4 gates
  g1[5] = (int)stride0;                          // tensor_dim0_stride [31:0]
  g1[6] = 0;                                     // stride high bits
  g1[7] = 0;
  i32x4 z4 = {0, 0, 0, 0};                       // groups 2/3 unused (2-D)
  i32x8 z8 = {0, 0, 0, 0, 0, 0, 0, 0};
  __builtin_amdgcn_tensor_load_to_lds(g0, g1, z4, z4, z8, 0);
}

// ---------------------------------------------------------------------------
// Pack a row-major f32 weight matrix W[N][K] into bf16 WMMA-B fragments.
// Tile = 16 N-cols x 32 K-rows = 512 bf16; hot loop reads v16bf at
// tile*512 + lane*16 (coalesced 1KB per wave / contiguous per gate slice).
// ---------------------------------------------------------------------------
__global__ __launch_bounds__(256)
void pack_b_kernel(const float* __restrict__ W, __bf16* __restrict__ out,
                   int N, int K) {
  size_t idx = (size_t)blockIdx.x * 256 + threadIdx.x;
  size_t total = (size_t)N * K;
  if (idx >= total) return;
  int tile = (int)(idx >> 9);
  int pos  = (int)(idx & 511);
  int lane = pos >> 4;
  int e    = pos & 15;
  int ktiles = K >> 5;
  int ntile = tile / ktiles;
  int ktile = tile - ntile * ktiles;
  int n = ntile * 16 + (lane & 15);
  int k = ktile * 32 + ((lane >> 4) << 3) + (e & 7) + ((e >> 3) << 4);
  out[idx] = (__bf16)W[(size_t)n * K + k];
}

// ---------------------------------------------------------------------------
// Embedding gather: f32 emb output (2nd reference output) + bf16 activations.
// ---------------------------------------------------------------------------
__global__ __launch_bounds__(128)
void embed_kernel(const int* __restrict__ tgt, const float* __restrict__ embt,
                  float* __restrict__ emb_out, __bf16* __restrict__ x0_bf) {
  int tb  = blockIdx.x;            // 0 .. T*B-1
  int tid = threadIdx.x;           // 128 threads, 4 floats each
  int token = tgt[tb];
  const float* src = embt + (size_t)token * IN_DIM;
  f32x4 v = *(const f32x4*)(src + tid * 4);
  *(f32x4*)(emb_out + (size_t)tb * IN_DIM + tid * 4) = v;
  size_t o = (size_t)tb * IN_DIM + tid * 4;
  x0_bf[o + 0] = (__bf16)v.x;
  x0_bf[o + 1] = (__bf16)v.y;
  x0_bf[o + 2] = (__bf16)v.z;
  x0_bf[o + 3] = (__bf16)v.w;
}

// ---------------------------------------------------------------------------
// Initialize recurrent state (slot 0) from h0/c0 ([1,L,B,H] f32).
// ---------------------------------------------------------------------------
__global__ __launch_bounds__(256)
void init_state_kernel(const float* __restrict__ h0, const float* __restrict__ c0,
                       __bf16* __restrict__ hbf_l0, __bf16* __restrict__ hbf_l1,
                       float* __restrict__ c_l0, float* __restrict__ c_l1) {
  int idx = blockIdx.x * 256 + threadIdx.x;   // 0 .. 2*64*1024-1
  int l = idx >> 16;
  int i = idx & 65535;
  float h = h0[idx], c = c0[idx];
  if (l == 0) { hbf_l0[i] = (__bf16)h; c_l0[i] = c; }
  else        { hbf_l1[i] = (__bf16)h; c_l1[i] = c; }
}

// ---------------------------------------------------------------------------
// LSTM cell: gates = x@Wih^T + h@Whh^T + b ; fused i/f/g/o nonlinearity.
// Grid: 64 blocks (one per hidden 16-col tile nh) x 128 threads (4 waves =
// 4 M-tiles). Wave 0 TDM-loads the block's weight slice to LDS:
//   smem[0      .. 64K)  : Wih slab, 4 gates x (16 kt x 512 bf16)
//   smem[65536 .. 192K)  : Whh slab, 4 gates x (32 kt x 512 bf16)
// hh slab streams while the x-part WMMAs run against the ih slab.
// ---------------------------------------------------------------------------
__global__ __launch_bounds__(128)
void lstm_cell_kernel(const __bf16* __restrict__ x_bf,    // [64][512]
                      const __bf16* __restrict__ h_bf,    // [64][1024]
                      const float*  __restrict__ c_prev,  // [64][1024]
                      const __bf16* __restrict__ wih_pk,  // packed (4*64 nt)x(16 kt)x512
                      const __bf16* __restrict__ whh_pk,  // packed (4*64 nt)x(32 kt)x512
                      const float*  __restrict__ b_ih,    // [4096]
                      const float*  __restrict__ b_hh,    // [4096]
                      float*  __restrict__ c_new,         // [64][1024]
                      __bf16* __restrict__ h_new_bf) {    // [64][1024]
  extern __shared__ char smem[];                // 196608 B dynamic, base off 0
  const int IH_KT = IN_DIM / 32;                // 16
  const int HH_KT = H_DIM / 32;                 // 32

  int nh   = blockIdx.x;                        // 0..63
  int wv   = threadIdx.x >> 5;                  // 0..3 == M tile
  int lane = threadIdx.x & 31;
  int mt   = wv;

  if (wv == 0) {
    // gate-g slice for this nh lives at ((g*64 + nh) * KT) * 512 bf16:
    // contiguous KT*512 bf16 per gate, gate stride 64*KT*512 bf16.
    const __bf16* ih_base = wih_pk + (size_t)nh * IH_KT * 512;
    const __bf16* hh_base = whh_pk + (size_t)nh * HH_KT * 512;
    tdm_load_2d(0u,     ih_base, IH_KT * 128u, 64u * IH_KT * 128u); // 8B elems
    tdm_load_2d(65536u, hh_base, HH_KT * 128u, 64u * HH_KT * 128u);
    __builtin_amdgcn_s_wait_tensorcnt(1);       // ih slab landed (in-order)
  }
  __syncthreads();

  v8f acc[4] = {v8f{}, v8f{}, v8f{}, v8f{}};

  // x contribution: K = 512, B fragments from LDS ih slab
  const char* lds_ih = smem;
  for (int kt = 0; kt < IH_KT; ++kt) {
    v16bf a = load_a_frag(x_bf, mt * 16 + (lane & 15), IN_DIM, kt * 32, lane);
#pragma unroll
    for (int g = 0; g < 4; ++g) {
      const v16bf* bp = (const v16bf*)(lds_ih +
          ((size_t)(g * IH_KT + kt) * 512 + lane * 16) * 2);
      acc[g] = __builtin_amdgcn_wmma_f32_16x16x32_bf16(
          false, a, false, *bp, (short)0, acc[g], false, false);
    }
  }

  if (wv == 0) __builtin_amdgcn_s_wait_tensorcnt(0);  // hh slab landed
  __syncthreads();

  // h contribution: K = 1024, B fragments from LDS hh slab
  const char* lds_hh = smem + 65536;
  for (int kt = 0; kt < HH_KT; ++kt) {
    v16bf a = load_a_frag(h_bf, mt * 16 + (lane & 15), H_DIM, kt * 32, lane);
#pragma unroll
    for (int g = 0; g < 4; ++g) {
      const v16bf* bp = (const v16bf*)(lds_hh +
          ((size_t)(g * HH_KT + kt) * 512 + lane * 16) * 2);
      acc[g] = __builtin_amdgcn_wmma_f32_16x16x32_bf16(
          false, a, false, *bp, (short)0, acc[g], false, false);
    }
  }

  // C/D layout: vgpr r, lanes 0-15 -> M=r, lanes 16-31 -> M=8+r, N=lane&15.
  int col = nh * 16 + (lane & 15);
  float bias[4];
#pragma unroll
  for (int g = 0; g < 4; ++g)
    bias[g] = b_ih[g * H_DIM + col] + b_hh[g * H_DIM + col];

  int rbase = mt * 16 + ((lane >> 4) << 3);
#pragma unroll
  for (int r = 0; r < 8; ++r) {
    int row = rbase + r;
    float ig = sigmoid_f(acc[0][r] + bias[0]);
    float fg = sigmoid_f(acc[1][r] + bias[1]);
    float gg = tanh_f   (acc[2][r] + bias[2]);
    float og = sigmoid_f(acc[3][r] + bias[3]);
    float c2 = fg * c_prev[(size_t)row * H_DIM + col] + ig * gg;
    float h2 = og * tanh_f(c2);
    c_new[(size_t)row * H_DIM + col] = c2;
    h_new_bf[(size_t)row * H_DIM + col] = (__bf16)h2;
  }
}

// ---------------------------------------------------------------------------
// Projection: out = h @ Wp^T + bp  ([64,1024] x [1024,512]).
// 128 waves (4 M-tiles x 32 N-tiles), direct L2-resident global B loads.
// ---------------------------------------------------------------------------
__global__ __launch_bounds__(256)
void proj_kernel(const __bf16* __restrict__ h_bf,     // [64][1024]
                 const __bf16* __restrict__ wp_pk,    // packed (32 nt)x(32 kt)x512
                 const float*  __restrict__ bp,       // [512]
                 float*  __restrict__ out_f32,        // [64][512] slice of d_out
                 __bf16* __restrict__ x_next) {       // [64][512] bf16 or null
  int wave = (blockIdx.x << 3) + (threadIdx.x >> 5);  // 0..127
  int lane = threadIdx.x & 31;
  int mt = wave & 3;
  int nt = wave >> 2;     // 0..31

  v8f acc = v8f{};
  for (int kt = 0; kt < H_DIM / 32; ++kt) {
    v16bf a = load_a_frag(h_bf, mt * 16 + (lane & 15), H_DIM, kt * 32, lane);
    size_t tile = (size_t)nt * (H_DIM / 32) + kt;
    v16bf b = *(const v16bf*)(wp_pk + tile * 512 + lane * 16);
    acc = __builtin_amdgcn_wmma_f32_16x16x32_bf16(
        false, a, false, b, (short)0, acc, false, false);
  }

  int col = nt * 16 + (lane & 15);
  float bias = bp[col];
  int rbase = mt * 16 + ((lane >> 4) << 3);
#pragma unroll
  for (int r = 0; r < 8; ++r) {
    int row = rbase + r;
    float v = acc[r] + bias;
    out_f32[(size_t)row * IN_DIM + col] = v;
    if (x_next) x_next[(size_t)row * IN_DIM + col] = (__bf16)v;
  }
}

// ---------------------------------------------------------------------------
// Host launch
// ---------------------------------------------------------------------------
extern "C" void kernel_launch(void* const* d_in, const int* in_sizes, int n_in,
                              void* d_out, int out_size, void* d_ws, size_t ws_size,
                              hipStream_t stream) {
  const int*   tgt  = (const int*)d_in[0];
  const float* h0   = (const float*)d_in[1];
  const float* c0   = (const float*)d_in[2];
  const float* embt = (const float*)d_in[3];
  const float* W_ih[2] = {(const float*)d_in[4],  (const float*)d_in[10]};
  const float* W_hh[2] = {(const float*)d_in[5],  (const float*)d_in[11]};
  const float* b_ih[2] = {(const float*)d_in[6],  (const float*)d_in[12]};
  const float* b_hh[2] = {(const float*)d_in[7],  (const float*)d_in[13]};
  const float* Wp[2]   = {(const float*)d_in[8],  (const float*)d_in[14]};
  const float* bp[2]   = {(const float*)d_in[9],  (const float*)d_in[15]};

  char* ws = (char*)d_ws;
  size_t off = 0;
  auto alloc = [&](size_t bytes) -> char* {
    char* p = ws + off;
    off += (bytes + 255) & ~(size_t)255;
    return p;
  };

  __bf16* wih_pk[2]; __bf16* whh_pk[2]; __bf16* wp_pk[2];
  for (int l = 0; l < 2; ++l) {
    wih_pk[l] = (__bf16*)alloc((size_t)4 * H_DIM * IN_DIM * 2);   // 4MB
    whh_pk[l] = (__bf16*)alloc((size_t)4 * H_DIM * H_DIM * 2);    // 8MB
    wp_pk[l]  = (__bf16*)alloc((size_t)IN_DIM * H_DIM * 2);       // 1MB
  }
  __bf16* x0_bf = (__bf16*)alloc((size_t)T_STEPS * BATCH * IN_DIM * 2);  // 16MB
  __bf16* x1_bf = (__bf16*)alloc((size_t)BATCH * IN_DIM * 2);
  __bf16* hbf[2][2];
  float*  cst[2][2];
  for (int l = 0; l < 2; ++l)
    for (int s = 0; s < 2; ++s) {
      hbf[l][s] = (__bf16*)alloc((size_t)BATCH * H_DIM * 2);
      cst[l][s] = (float*)alloc((size_t)BATCH * H_DIM * 4);
    }

  // d_out: [1,T,L,B,IN] f32 followed by [1,T,B,IN] f32
  float* dir_out = (float*)d_out;
  float* emb_out = dir_out + (size_t)T_STEPS * 2 * BATCH * IN_DIM;

  // --- one-time: pack weights to bf16 fragment order ---
  for (int l = 0; l < 2; ++l) {
    size_t n;
    n = (size_t)4 * H_DIM * IN_DIM;
    pack_b_kernel<<<(unsigned)((n + 255) / 256), 256, 0, stream>>>(W_ih[l], wih_pk[l], 4 * H_DIM, IN_DIM);
    n = (size_t)4 * H_DIM * H_DIM;
    pack_b_kernel<<<(unsigned)((n + 255) / 256), 256, 0, stream>>>(W_hh[l], whh_pk[l], 4 * H_DIM, H_DIM);
    n = (size_t)IN_DIM * H_DIM;
    pack_b_kernel<<<(unsigned)((n + 255) / 256), 256, 0, stream>>>(Wp[l], wp_pk[l], IN_DIM, H_DIM);
  }

  // --- one-time: embedding gather + bf16 activations ---
  embed_kernel<<<T_STEPS * BATCH, 128, 0, stream>>>(tgt, embt, emb_out, x0_bf);

  // --- one-time: initial state (slot 0) ---
  init_state_kernel<<<512, 256, 0, stream>>>(h0, c0, hbf[0][0], hbf[1][0],
                                             cst[0][0], cst[1][0]);

  const size_t CELL_LDS = 196608;   // 64KB ih slab + 128KB hh slab

  // --- recurrent loop: t = 0..255, 2 layers, cell + projection ---
  for (int t = 0; t < T_STEPS; ++t) {
    int rs = t & 1, wsl = 1 - rs;
    const __bf16* x_l0 = x0_bf + (size_t)t * BATCH * IN_DIM;

    // layer 0
    lstm_cell_kernel<<<64, 128, CELL_LDS, stream>>>(
        x_l0, hbf[0][rs], cst[0][rs], wih_pk[0], whh_pk[0],
        b_ih[0], b_hh[0], cst[0][wsl], hbf[0][wsl]);
    proj_kernel<<<16, 256, 0, stream>>>(
        hbf[0][wsl], wp_pk[0], bp[0],
        dir_out + ((size_t)t * 2 + 0) * BATCH * IN_DIM, x1_bf);

    // layer 1
    lstm_cell_kernel<<<64, 128, CELL_LDS, stream>>>(
        x1_bf, hbf[1][rs], cst[1][rs], wih_pk[1], whh_pk[1],
        b_ih[1], b_hh[1], cst[1][wsl], hbf[1][wsl]);
    proj_kernel<<<16, 256, 0, stream>>>(
        hbf[1][wsl], wp_pk[1], bp[1],
        dir_out + ((size_t)t * 2 + 1) * BATCH * IN_DIM, (__bf16*)nullptr);
  }
}
